// scaled_dot_77549929497146
// MI455X (gfx1250) — matrix operand order repaired and checked by reading, not compiled, not run on hardware
//
#include <hip/hip_runtime.h>

// ---------------------------------------------------------------------------
// Problem constants (from reference): B=8, SQ=2048, SK=2048, D=128
// out[b,q,d] = sum_k softmax_over_b(Q·K^T/128)[b,q,k] * V[b,k,d]
// ---------------------------------------------------------------------------
#define BN  8
#define SQN 2048
#define SKN 2048
#define DN  128

typedef __attribute__((ext_vector_type(16))) __bf16 v16bf;
typedef __attribute__((ext_vector_type(8)))  float  v8f;
typedef __attribute__((ext_vector_type(4)))  int    i4;
typedef __attribute__((ext_vector_type(2)))  float  f2;

union FragU { v16bf bf; i4 i[2]; };

__device__ __forceinline__ unsigned short f2bf(float x) {
    // round-to-nearest-even fp32 -> bf16
    unsigned int u = __builtin_bit_cast(unsigned int, x);
    unsigned int r = u + 0x7FFFu + ((u >> 16) & 1u);
    return (unsigned short)(r >> 16);
}

// ---------------------------------------------------------------------------
// Prep kernel 1: fp32 -> bf16, 2 elements per thread (packed dword stores)
// ---------------------------------------------------------------------------
__global__ void cvt_bf16x2_kernel(const float* __restrict__ src,
                                  unsigned int* __restrict__ dst, int npairs) {
    int i = blockIdx.x * blockDim.x + threadIdx.x;
    if (i < npairs) {
        f2 v = *(const f2*)(src + 2 * (size_t)i);
        dst[i] = (unsigned int)f2bf(v.x) | ((unsigned int)f2bf(v.y) << 16);
    }
}

// ---------------------------------------------------------------------------
// Prep kernel 2: Vt[b,d,k] = bf16(V[b,k,d])   (32x32 LDS tiled transpose)
// ---------------------------------------------------------------------------
__global__ void transpose_v_kernel(const float* __restrict__ V,
                                   unsigned short* __restrict__ Vt) {
    __shared__ float tile[32][33];
    const int b  = blockIdx.z;
    const int k0 = blockIdx.x * 32;
    const int d0 = blockIdx.y * 32;
#pragma unroll
    for (int r = 0; r < 32; r += 8)
        tile[threadIdx.y + r][threadIdx.x] =
            V[((size_t)(b * SKN + k0 + threadIdx.y + r)) * DN + d0 + threadIdx.x];
    __syncthreads();
#pragma unroll
    for (int r = 0; r < 32; r += 8) {
        int d = d0 + threadIdx.y + r;
        int k = k0 + threadIdx.x;
        Vt[((size_t)(b * DN + d)) * SKN + k] = f2bf(tile[threadIdx.x][threadIdx.y + r]);
    }
}

// ---------------------------------------------------------------------------
// Main kernel: 8 waves per WG, wave b owns batch b; WG owns a 16-row q tile.
// Per 32-wide k tile: 8 WMMAs (scores) + LDS batch-softmax + 8 WMMAs (P·V).
// ---------------------------------------------------------------------------
__launch_bounds__(256)
__global__ void attn_bsoftmax_kernel(const unsigned short* __restrict__ Qb,
                                     const unsigned short* __restrict__ Kb,
                                     const unsigned short* __restrict__ Vt,
                                     float* __restrict__ Out) {
    __shared__ __align__(16) float        ldsE[BN * 512];   // exp tiles, fp32 [b][16*32]
    __shared__ __align__(16) unsigned int ldsP[BN * 256];   // normalized bf16 P [b][16*32]

    const int tid  = threadIdx.x;
    const int b    = tid >> 5;          // wave id == batch id
    const int lane = tid & 31;
    const int half = lane >> 4;
    const int l16  = lane & 15;
    const int q0   = blockIdx.x * 16;

    // --- resident Q A-fragments: 4 chunks of K(d)=32, 16x32 bf16 each ------
    FragU qf[4];
    {
        const unsigned short* qrow = Qb + ((size_t)(b * SQN + q0 + l16)) * DN;
#pragma unroll
        for (int c = 0; c < 4; c++) {
            qf[c].i[0] = *(const i4*)(qrow + c * 32 + half * 8);
            qf[c].i[1] = *(const i4*)(qrow + c * 32 + 16 + half * 8);
        }
    }

    // --- resident O accumulator: 16(q) x 128(d) fp32 = 8 C-tiles ----------
    v8f acc[8];
#pragma unroll
    for (int i = 0; i < 8; i++) acc[i] = (v8f){};

    for (int kt = 0; kt < SKN / 32; kt++) {
        const int k0 = kt * 32;

        // ---- scores S = Q · K^T / 128, then exp -> LDS --------------------
#pragma unroll
        for (int nsub = 0; nsub < 2; nsub++) {
            v8f s = (v8f){};
            const int kcol = k0 + nsub * 16 + l16;   // this lane's key column
            const unsigned short* krow = Kb + ((size_t)(b * SKN + kcol)) * DN;
#pragma unroll
            for (int c = 0; c < 4; c++) {
                FragU kf;  // B-fragment 32(d) x 16(k): lanes=col, half picks d-range
                kf.i[0] = *(const i4*)(krow + c * 32 + half * 8);
                kf.i[1] = *(const i4*)(krow + c * 32 + 16 + half * 8);
                s = __builtin_amdgcn_wmma_f32_16x16x32_bf16(
                        false, qf[c].bf, false, kf.bf, (short)0, s, false, false);
            }
#pragma unroll
            for (int v = 0; v < 8; v++) {
                float e = __expf(s[v] * (1.0f / 128.0f));
                int   m = v + half * 8;              // C-layout row
                ldsE[b * 512 + m * 32 + nsub * 16 + l16] = e;
            }
        }
        __syncthreads();

        // ---- batch softmax: each of 256 threads owns 2 adjacent elements --
        {
            f2 ev[BN];
            float s0 = 0.f, s1 = 0.f;
#pragma unroll
            for (int bb = 0; bb < BN; bb++) {
                ev[bb] = *(const f2*)&ldsE[bb * 512 + tid * 2];
                s0 += ev[bb].x; s1 += ev[bb].y;
            }
            const float r0 = 1.0f / s0, r1 = 1.0f / s1;
#pragma unroll
            for (int bb = 0; bb < BN; bb++) {
                unsigned int pk = (unsigned int)f2bf(ev[bb].x * r0) |
                                  ((unsigned int)f2bf(ev[bb].y * r1) << 16);
                ldsP[bb * 256 + tid] = pk;           // row-major bf16 [16][32]
            }
        }
        __syncthreads();

        // ---- reload P as 16x32 bf16 A-fragment ----------------------------
        FragU pf;
        {
            const unsigned short* prow = (const unsigned short*)&ldsP[b * 256];
            pf.i[0] = *(const i4*)(prow + l16 * 32 + half * 8);
            pf.i[1] = *(const i4*)(prow + l16 * 32 + 16 + half * 8);
        }

        // ---- O += P · V  (V pre-transposed: rows k contiguous) ------------
#pragma unroll
        for (int dsub = 0; dsub < 8; dsub++) {
            FragU vf;
            const int dcol = dsub * 16 + l16;
            const unsigned short* vrow =
                Vt + ((size_t)(b * DN + dcol)) * SKN + k0 + half * 8;
            vf.i[0] = *(const i4*)(vrow);
            vf.i[1] = *(const i4*)(vrow + 16);
            acc[dsub] = __builtin_amdgcn_wmma_f32_16x16x32_bf16(
                            false, pf.bf, false, vf.bf, (short)0, acc[dsub], false, false);
        }
    }

    // ---- write O (fp32, C-layout -> coalesced b32 stores) -----------------
#pragma unroll
    for (int dsub = 0; dsub < 8; dsub++) {
#pragma unroll
        for (int v = 0; v < 8; v++) {
            int m = v + half * 8;
            Out[((size_t)(b * SQN + q0 + m)) * DN + dsub * 16 + l16] = acc[dsub][v];
        }
    }
}

// ---------------------------------------------------------------------------
extern "C" void kernel_launch(void* const* d_in, const int* in_sizes, int n_in,
                              void* d_out, int out_size, void* d_ws, size_t ws_size,
                              hipStream_t stream) {
    const float* Q = (const float*)d_in[0];
    const float* K = (const float*)d_in[1];
    const float* V = (const float*)d_in[2];
    float* Out = (float*)d_out;

    const size_t nElem = (size_t)BN * SQN * DN;      // 2,097,152 per tensor
    unsigned short* Qb = (unsigned short*)d_ws;
    unsigned short* Kb = Qb + nElem;
    unsigned short* Vt = Kb + nElem;

    const int npairs = (int)(nElem / 2);
    const int cvtBlocks = (npairs + 255) / 256;
    cvt_bf16x2_kernel<<<cvtBlocks, 256, 0, stream>>>(Q, (unsigned int*)Qb, npairs);
    cvt_bf16x2_kernel<<<cvtBlocks, 256, 0, stream>>>(K, (unsigned int*)Kb, npairs);
    transpose_v_kernel<<<dim3(SKN / 32, DN / 32, BN), dim3(32, 8), 0, stream>>>(V, Vt);

    attn_bsoftmax_kernel<<<SQN / 16, 256, 0, stream>>>(Qb, Kb, Vt, Out);
}
